// DilatedResidualBlock_34849364640183
// MI455X (gfx1250) — matrix operand order repaired and checked by reading, not compiled, not run on hardware
//
#include <hip/hip_runtime.h>
#include <hip/hip_bf16.h>
#include <math.h>

typedef __attribute__((ext_vector_type(16))) _Float16 v16h;
typedef __attribute__((ext_vector_type(8)))  float    v8f;

#define KK 16

__device__ __forceinline__ float lrelu(float v) { return v > 0.f ? v : 0.2f * v; }

// ---------------------------------------------------------------------------
// h0 = lrelu(x @ W_start + b_start)   [N,32] = [N,32]@[32,32]
// ---------------------------------------------------------------------------
__global__ __launch_bounds__(256) void start_kernel(
    const float* __restrict__ x, const float* __restrict__ W,
    const float* __restrict__ b, float* __restrict__ h0, int n)
{
    int idx = blockIdx.x * blockDim.x + threadIdx.x;
    if (idx >= n * 32) return;
    int i = idx >> 5, c = idx & 31;
    float acc = b[c];
    #pragma unroll
    for (int k = 0; k < 32; ++k) acc += x[i * 32 + k] * W[k * 32 + c];
    h0[idx] = lrelu(acc);
}

// ---------------------------------------------------------------------------
// One RandLA attention conv. One wave per node-iteration; 8 waves per block;
// grid-stride loop over nodes so the W_att B-operand register preload and all
// LDS weight staging amortize over ~12 nodes per wave.
// ---------------------------------------------------------------------------
__global__ __launch_bounds__(256) void conv_kernel(
    const float* __restrict__ hin,   // [n,32]
    const float* __restrict__ pos,   // [n,3]
    const int*   __restrict__ src,   // [n*16]
    const float* __restrict__ Wr,    // [10,32]
    const float* __restrict__ br,    // [32]
    const float* __restrict__ Wa,    // [64,64]
    const float* __restrict__ Wp,    // [64,dpost]
    const float* __restrict__ bp,    // [dpost]
    float* __restrict__ hout,        // [n,dpost]
    int n, int dpost)
{
    __shared__ _Float16 satt[64 * 64];      // W_att as f16          (8 KB)
    __shared__ float    swr[10 * 32];       // W_rppe                (1.25 KB)
    __shared__ float    sbr[32];
    __shared__ float    swp[64 * 64];       // W_post (max 64x64)    (16 KB)
    __shared__ float    sbp[64];
    __shared__ _Float16 sfhat[8][16 * 64];  // per-wave f_hat tile   (16 KB)
    __shared__ float    sout[8][64];        // per-wave pooled vec   (2 KB)

    const int tid = threadIdx.x;
    for (int idx = tid; idx < 64 * 64; idx += 256) satt[idx] = (_Float16)Wa[idx];
    for (int idx = tid; idx < 10 * 32; idx += 256) swr[idx] = Wr[idx];
    for (int idx = tid; idx < 64 * dpost; idx += 256) swp[idx] = Wp[idx];
    if (tid < 32)    sbr[tid] = br[tid];
    if (tid < dpost) sbp[tid] = bp[tid];
    __syncthreads();

    const int wave = tid >> 5;
    const int lane = tid & 31;
    const int j    = lane & 15;   // edge index within node / local N column
    const int hl   = lane >> 4;   // lane half

    // ---- preload B operands of W_att once (constant over all nodes) ----
    // B layout (32x16 f16): lanes 0-15 hold K=0..7,16..23 of column N=lane,
    // lanes 16-31 hold K=8..15,24..31 of column N=lane-16 (matches A order).
    v16h B[4][2];
    #pragma unroll
    for (int t = 0; t < 4; ++t) {
        #pragma unroll
        for (int h = 0; h < 2; ++h) {
            const int ncol  = t * 16 + j;
            const int kbase = h * 32 + hl * 8;
            v16h bb;
            #pragma unroll
            for (int p = 0; p < 16; ++p)
                bb[p] = satt[(kbase + p + (p >> 3) * 8) * 64 + ncol];
            B[t][h] = bb;
        }
    }

    _Float16* fh = &sfhat[wave][0];
    const int stride = gridDim.x * 8;

    for (int node = blockIdx.x * 8 + wave; node < n; node += stride) {
        // ---- per-edge relative-position encoding + gather; edge = j ----
        const int   sj  = src[node * KK + j];
        const float pix = pos[node * 3 + 0], piy = pos[node * 3 + 1], piz = pos[node * 3 + 2];
        const float pjx = pos[sj * 3 + 0],  pjy = pos[sj * 3 + 1],  pjz = pos[sj * 3 + 2];
        const float dx = pix - pjx, dy = piy - pjy, dz = piz - pjz;
        const float dist = sqrtf(dx * dx + dy * dy + dz * dz);
        const float rp[10] = {pix, piy, piz, pjx, pjy, pjz, dx, dy, dz, dist};

        // lane (j,hl) fills channels hl*16..hl*16+15 of edge j:
        //   fh[j][c]    = hin[src][c]          (gather half of f_hat)
        //   fh[j][32+c] = lrelu(rp @ Wr + br)  (rppe half)
        #pragma unroll
        for (int p = 0; p < 16; ++p) {
            const int c = hl * 16 + p;
            float acc = sbr[c];
            #pragma unroll
            for (int q = 0; q < 10; ++q) acc += rp[q] * swr[q * 32 + c];
            fh[j * 64 + 32 + c] = (_Float16)lrelu(acc);
            fh[j * 64 + c]      = (_Float16)hin[sj * 32 + c];
        }

        // wave-internal LDS handoff (DS ops are in-order per wave)
        asm volatile("s_wait_dscnt 0" ::: "memory");

        // ---- A operands in documented 16-bit A layout (M=16, K=32 each) ----
        const int row  = j;
        const int koff = hl * 8;
        v16h A0, A1;
        #pragma unroll
        for (int p = 0; p < 8; ++p) {
            A0[p]     = fh[row * 64 +      koff + p];
            A0[8 + p] = fh[row * 64 + 16 + koff + p];
            A1[p]     = fh[row * 64 + 32 + koff + p];
            A1[8 + p] = fh[row * 64 + 48 + koff + p];
        }

        // ---- logits = f_hat @ W_att, fused softmax + weighted pooling ----
        #pragma unroll
        for (int t = 0; t < 4; ++t) {
            v8f c = {};
            c = __builtin_amdgcn_wmma_f32_16x16x32_f16(false, A0, false, B[t][0],
                                                       (short)0, c, false, false);
            c = __builtin_amdgcn_wmma_f32_16x16x32_f16(false, A1, false, B[t][1],
                                                       (short)0, c, false, false);
            // C layout: lane holds channel ch = t*16 + j, edges M = hl*8 + p
            float mx = c[0];
            #pragma unroll
            for (int p = 1; p < 8; ++p) mx = fmaxf(mx, c[p]);
            mx = fmaxf(mx, __shfl_xor(mx, 16));
            float e[8], s = 0.f;
            #pragma unroll
            for (int p = 0; p < 8; ++p) { e[p] = __expf(c[p] - mx); s += e[p]; }
            s += __shfl_xor(s, 16);
            const float inv = 1.f / s;
            const int   ch  = t * 16 + j;
            float acc = 0.f;
            #pragma unroll
            for (int p = 0; p < 8; ++p) {
                const int m = hl * 8 + p;
                acc += (e[p] * inv) * (float)fh[m * 64 + ch];
            }
            acc += __shfl_xor(acc, 16);
            if (hl == 0) sout[wave][ch] = acc;
        }

        asm volatile("s_wait_dscnt 0" ::: "memory");

        // ---- post MLP: hout = lrelu(out[64] @ Wp + bp) ----
        for (int co = lane; co < dpost; co += 32) {
            float acc = sbp[co];
            #pragma unroll 8
            for (int k = 0; k < 64; ++k)
                acc += sout[wave][k] * swp[k * dpost + co];
            hout[node * dpost + co] = lrelu(acc);
        }
    }
}

// ---------------------------------------------------------------------------
// out = lrelu(h2 @ W_end + b_end + x @ W_skip + b_skip)  — one block per node
// ---------------------------------------------------------------------------
__global__ __launch_bounds__(128) void end_kernel(
    const float* __restrict__ h2, const float* __restrict__ x,
    const float* __restrict__ We, const float* __restrict__ be,
    const float* __restrict__ Ws, const float* __restrict__ bs,
    float* __restrict__ out, int n)
{
    __shared__ float sh[64], sx[32];
    const int i = blockIdx.x;
    const int c = threadIdx.x;
    if (c < 64) sh[c] = h2[(size_t)i * 64 + c];
    if (c < 32) sx[c] = x[(size_t)i * 32 + c];
    __syncthreads();
    float acc = be[c] + bs[c];
    #pragma unroll 8
    for (int k = 0; k < 64; ++k) acc += sh[k] * We[k * 128 + c];
    #pragma unroll 8
    for (int k = 0; k < 32; ++k) acc += sx[k] * Ws[k * 128 + c];
    out[(size_t)i * 128 + c] = lrelu(acc);
}

// ---------------------------------------------------------------------------
extern "C" void kernel_launch(void* const* d_in, const int* in_sizes, int n_in,
                              void* d_out, int out_size, void* d_ws, size_t ws_size,
                              hipStream_t stream) {
    const float* x       = (const float*)d_in[0];
    const float* pos     = (const float*)d_in[1];
    const float* W_start = (const float*)d_in[2];
    const float* b_start = (const float*)d_in[3];
    const float* W_skip  = (const float*)d_in[4];
    const float* b_skip  = (const float*)d_in[5];
    const float* W_rppe1 = (const float*)d_in[6];
    const float* b_rppe1 = (const float*)d_in[7];
    const float* W_att1  = (const float*)d_in[8];
    const float* W_post1 = (const float*)d_in[9];
    const float* b_post1 = (const float*)d_in[10];
    const float* W_rppe2 = (const float*)d_in[11];
    const float* b_rppe2 = (const float*)d_in[12];
    const float* W_att2  = (const float*)d_in[13];
    const float* W_post2 = (const float*)d_in[14];
    const float* b_post2 = (const float*)d_in[15];
    const float* W_end   = (const float*)d_in[16];
    const float* b_end   = (const float*)d_in[17];
    const int*   ei      = (const int*)d_in[18];   // int32 (JAX x64 disabled)

    const int n = in_sizes[0] / 32;   // N
    const int* srcIdx = ei;           // edge_index[0] = src, row-contiguous

    float* h0 = (float*)d_ws;                 // [n,32]
    float* h1 = h0 + (size_t)n * 32;          // [n,32]
    float* h2 = h1 + (size_t)n * 32;          // [n,64]

    int convBlocks = (n + 7) / 8;
    if (convBlocks > 1024) convBlocks = 1024;   // grid-stride, ~12 nodes/wave

    start_kernel<<<(n * 32 + 255) / 256, 256, 0, stream>>>(x, W_start, b_start, h0, n);
    conv_kernel<<<convBlocks, 256, 0, stream>>>(h0, pos, srcIdx, W_rppe1, b_rppe1,
                                                W_att1, W_post1, b_post1, h1, n, 32);
    conv_kernel<<<convBlocks, 256, 0, stream>>>(h1, pos, srcIdx, W_rppe2, b_rppe2,
                                                W_att2, W_post2, b_post2, h2, n, 64);
    end_kernel<<<n, 128, 0, stream>>>(h2, x, W_end, b_end, W_skip, b_skip,
                                      (float*)d_out, n);
}
